// MoE_75170517615144
// MI455X (gfx1250) — compile-verified, matrix operand order checked by reading, not run on hardware
//
#include <hip/hip_runtime.h>
#include <hip/hip_bf16.h>
#include <stdint.h>

#define DIM 1024
#define INTER 1024
#define NEXP 16
#define TOPK 4
#define SH_INTER 2048
#define NTOK 8192

#define KS 64                 // K elements staged per LDS stage
#define PITCH 72              // bf16 elements per staged row (64 + 8 pad = 144B)
#define TILE_E (128 * PITCH)  // elements per 128-row tile

typedef __bf16 bf16_t;
typedef __attribute__((ext_vector_type(16))) __bf16 bf16x16;
typedef __attribute__((ext_vector_type(8)))  __bf16 bf16x8;
typedef __attribute__((ext_vector_type(4)))  __bf16 bf16x4;
typedef __attribute__((ext_vector_type(8)))  float  f32x8;
typedef __attribute__((ext_vector_type(4)))  int    int4v;

typedef __attribute__((address_space(1))) int4v gas_int4;  // global
typedef __attribute__((address_space(3))) int4v las_int4;  // LDS

#if defined(__has_builtin)
#if __has_builtin(__builtin_amdgcn_global_load_async_to_lds_b128)
#define HAVE_ASYNC_LDS 1
#endif
#endif
#ifndef HAVE_ASYNC_LDS
#define HAVE_ASYNC_LDS 0
#endif

// ---------------------------------------------------------------------------
// Fragment: CDNA5 16-bit A/B operand layout (ISA 7.12.2).
// Lane L holds row (L&15); K elements {koff..koff+7, koff+16..koff+23},
// koff = (L>=16)?8:0. Two contiguous 16B chunks per fragment.
// ---------------------------------------------------------------------------
static __device__ __forceinline__ bf16x16 load_frag(const bf16_t* p) {
  bf16x8 lo = *(const bf16x8*)(p);
  bf16x8 hi = *(const bf16x8*)(p + 16);
  return __builtin_shufflevector(lo, hi,
      0, 1, 2, 3, 4, 5, 6, 7, 8, 9, 10, 11, 12, 13, 14, 15);
}

static __device__ __forceinline__ f32x8 wmma_bf16(bf16x16 a, bf16x16 b, f32x8 c) {
  return __builtin_amdgcn_wmma_f32_16x16x32_bf16(
      false, a, false, b, (short)0, c, false, false);
}

// Stage one 16B chunk: global -> LDS (ASYNCcnt-tracked when available)
static __device__ __forceinline__ void stage_chunk(const bf16_t* g, bf16_t* l) {
#if HAVE_ASYNC_LDS
  __builtin_amdgcn_global_load_async_to_lds_b128(
      (gas_int4*)g, (las_int4*)l, 0, 0);
#else
  *(bf16x8*)l = *(const bf16x8*)g;
#endif
}

static __device__ __forceinline__ void wait_async() {
#if HAVE_ASYNC_LDS
#if __has_builtin(__builtin_amdgcn_s_wait_asynccnt)
  __builtin_amdgcn_s_wait_asynccnt(0);
#else
  asm volatile("s_wait_asynccnt 0" ::: "memory");
#endif
#endif
}

// ---------------------------------------------------------------------------
// fp32 -> bf16 conversion
// ---------------------------------------------------------------------------
__global__ void cvt_bf16_kernel(const float* __restrict__ src,
                                bf16_t* __restrict__ dst, int n) {
  int i = (blockIdx.x * blockDim.x + threadIdx.x) * 4;
  if (i + 3 < n) {
    float4 f = *(const float4*)(src + i);
    bf16x4 o;
    o[0] = (bf16_t)f.x; o[1] = (bf16_t)f.y;
    o[2] = (bf16_t)f.z; o[3] = (bf16_t)f.w;
    *(bf16x4*)(dst + i) = o;
  } else {
    for (; i < n; ++i) dst[i] = (bf16_t)src[i];
  }
}

// ---------------------------------------------------------------------------
// Gate: softmax(x @ gate_w.T), top-4, dispatch lists. One wave per token.
// Dense per-(token,k) row id = t*TOPK + k  (bit-deterministic combine).
// ---------------------------------------------------------------------------
__global__ void gate_topk_kernel(const float* __restrict__ x,
                                 const float* __restrict__ gw,
                                 int* __restrict__ cnt,
                                 int* __restrict__ tok,
                                 int* __restrict__ hrow,
                                 float* __restrict__ wgt) {
  int wave = threadIdx.x >> 5;
  int lane = threadIdx.x & 31;
  int t = blockIdx.x * (blockDim.x >> 5) + wave;
  if (t >= NTOK) return;

  const float* xr = x + (size_t)t * DIM;
  float sc[NEXP];
#pragma unroll
  for (int e = 0; e < NEXP; ++e) {
    const float* gr = gw + (size_t)e * DIM;
    float p = 0.f;
    for (int i = lane; i < DIM; i += 32) p += xr[i] * gr[i];
#pragma unroll
    for (int off = 16; off >= 1; off >>= 1) p += __shfl_xor(p, off, 32);
    sc[e] = p;
  }
  float m = sc[0];
#pragma unroll
  for (int e = 1; e < NEXP; ++e) m = fmaxf(m, sc[e]);
  float s = 0.f;
#pragma unroll
  for (int e = 0; e < NEXP; ++e) { sc[e] = __expf(sc[e] - m); s += sc[e]; }
  float inv = 1.f / s;
#pragma unroll
  for (int e = 0; e < NEXP; ++e) sc[e] *= inv;

  if (lane == 0) {
#pragma unroll
    for (int k = 0; k < TOPK; ++k) {
      int be = 0; float bv = sc[0];
#pragma unroll
      for (int e = 1; e < NEXP; ++e)
        if (sc[e] > bv) { bv = sc[e]; be = e; }   // first-index tie-break
      sc[be] = -1.f;
      int r = atomicAdd(&cnt[be], 1);
      tok[be * NTOK + r]  = t;
      hrow[be * NTOK + r] = t * TOPK + k;
      wgt[be * NTOK + r]  = bv;
    }
  }
}

// ---------------------------------------------------------------------------
// Fused gate/up GEMM, double-buffered LDS staging:
//   H = silu(A @ W1^T + b1) * (A @ W3^T + b3)  (bf16 out)
// Block tile 128(M) x 128(N), 8 waves 2(M)x4(N), wave tile 64x32.
// LDS: 2 x (A[128xKS] + W1[128xKS] + W3[128xKS]); copy(s+1) overlaps wmma(s).
// ---------------------------------------------------------------------------
__global__ __launch_bounds__(256) void gemm_gate_up_kernel(
    const bf16_t* __restrict__ X,    // [*, K]
    const bf16_t* __restrict__ W1,   // [E, N, K] or [N, K]
    const bf16_t* __restrict__ W3,
    const float* __restrict__ B1,
    const float* __restrict__ B3,
    bf16_t* __restrict__ H,          // [*, ldh]
    const int* __restrict__ tok,     // nullptr => shared expert
    const int* __restrict__ hrow,
    const int* __restrict__ cnt,
    int N, int K, int ldh) {
  __shared__ bf16_t sm[2 * 3 * TILE_E];  // 110592 B (of 320KB WGP LDS)

  const int e = blockIdx.z;
  const int rows = tok ? cnt[e] : NTOK;
  const int rowBase = blockIdx.y * 128;
  if (rowBase >= rows) return;
  const int colBlock = blockIdx.x * 128;

  const bf16_t* w1 = W1 + (size_t)e * N * K;
  const bf16_t* w3 = W3 + (size_t)e * N * K;
  const float*  b1 = B1 + (size_t)e * N;
  const float*  b3 = B3 + (size_t)e * N;
  const int* tokE  = tok  ? tok  + (size_t)e * NTOK : nullptr;
  const int* hrowE = hrow ? hrow + (size_t)e * NTOK : nullptr;

  const int tid  = threadIdx.x;
  const int wave = tid >> 5;
  const int lane = tid & 31;
  const int wm = wave & 1;        // 2 waves over M (64 rows each)
  const int wn = wave >> 1;       // 4 waves over N (32 cols each)
  const int l15 = lane & 15;
  const int koff = (lane >> 4) << 3;

  // staging assignment: 12 chunks of 16B per thread per stage
  // chunk id c = i*256 + tid ; tile = c>>10 (0=A,1=W1,2=W3), r=(c>>3)&127, ck=c&7
  const bf16_t* gsrc[12];
  int loff[12];
#pragma unroll
  for (int i = 0; i < 12; ++i) {
    int c = i * 256 + tid;
    int tile = c >> 10;
    int r = (c >> 3) & 127;
    int ck = c & 7;
    const bf16_t* base;
    if (tile == 0) {
      int lr = rowBase + r;
      int ar;
      if (tokE) ar = (lr < rows) ? tokE[lr] : 0;
      else      ar = (lr < rows) ? lr : 0;
      base = X + (size_t)ar * K;
    } else {
      int col = colBlock + r;
      base = (tile == 1 ? w1 : w3) + (size_t)col * K;
    }
    gsrc[i] = base + ck * 8;
    loff[i] = tile * TILE_E + r * PITCH + ck * 8;
  }

  f32x8 accG[4][2] = {};
  f32x8 accU[4][2] = {};

  // prologue: stage buffer 0
#pragma unroll
  for (int i = 0; i < 12; ++i) stage_chunk(gsrc[i], &sm[loff[i]]);

  const int nst = K / KS;
#pragma unroll 1
  for (int s = 0; s < nst; ++s) {
    wait_async();
    __syncthreads();
    if (s + 1 < nst) {           // overlap next stage's DMA with this wmma block
      bf16_t* dst = &sm[((s + 1) & 1) * 3 * TILE_E];
      int k1 = (s + 1) * KS;
#pragma unroll
      for (int i = 0; i < 12; ++i) stage_chunk(gsrc[i] + k1, dst + loff[i]);
    }
    const bf16_t* cur = &sm[(s & 1) * 3 * TILE_E];

#pragma unroll
    for (int kk = 0; kk < KS; kk += 32) {
      bf16x16 af[4];
#pragma unroll
      for (int mi = 0; mi < 4; ++mi) {
        int row = wm * 64 + mi * 16 + l15;
        af[mi] = load_frag(&cur[row * PITCH + kk + koff]);
      }
#pragma unroll
      for (int ni = 0; ni < 2; ++ni) {
        int cl = wn * 32 + ni * 16 + l15;
        bf16x16 bg = load_frag(&cur[TILE_E     + cl * PITCH + kk + koff]);
        bf16x16 bu = load_frag(&cur[2 * TILE_E + cl * PITCH + kk + koff]);
#pragma unroll
        for (int mi = 0; mi < 4; ++mi) {
          accG[mi][ni] = wmma_bf16(af[mi], bg, accG[mi][ni]);
          accU[mi][ni] = wmma_bf16(af[mi], bu, accU[mi][ni]);
        }
      }
    }
  }

  // Epilogue: C/D layout: VGPR v -> row v + (lane>=16 ? 8:0), col = lane&15.
  const int rsel = (lane >= 16) ? 8 : 0;
#pragma unroll
  for (int mi = 0; mi < 4; ++mi) {
    int lrbase = rowBase + wm * 64 + mi * 16;
#pragma unroll
    for (int ni = 0; ni < 2; ++ni) {
      int col = colBlock + wn * 32 + ni * 16 + l15;
      float bb1 = b1[col];
      float bb3 = b3[col];
#pragma unroll
      for (int v = 0; v < 8; ++v) {
        int lr = lrbase + v + rsel;
        if (lr < rows) {
          float g = accG[mi][ni][v] + bb1;
          float u = accU[mi][ni][v] + bb3;
          float h = (g / (1.f + __expf(-g))) * u;   // silu(g) * u
          int dr = tokE ? hrowE[lr] : lr;
          H[(size_t)dr * ldh + col] = (bf16_t)h;
        }
      }
    }
  }
}

// ---------------------------------------------------------------------------
// Down GEMM, double-buffered LDS staging: Out = scale * (A @ W2^T + b2)
// routed: A rows / out rows = dense (t*4+k), scale = wgt ; shared: identity.
// ---------------------------------------------------------------------------
__global__ __launch_bounds__(256) void gemm_down_kernel(
    const bf16_t* __restrict__ Hsrc,  // [*, K]
    const bf16_t* __restrict__ W2,    // [E, N, K] or [N, K]
    const float* __restrict__ B2,
    float* __restrict__ Out,          // yr [NTOK*TOPK, N] or d_out [NTOK, N]
    const int* __restrict__ hrow,     // nullptr => shared
    const float* __restrict__ wgt,
    const int* __restrict__ cnt,
    int N, int K) {
  __shared__ bf16_t sm[2 * 2 * TILE_E];  // 73728 B

  const int e = blockIdx.z;
  const int rows = hrow ? cnt[e] : NTOK;
  const int rowBase = blockIdx.y * 128;
  if (rowBase >= rows) return;
  const int colBlock = blockIdx.x * 128;

  const bf16_t* w2 = W2 + (size_t)e * N * K;
  const float*  b2 = B2 + (size_t)e * N;
  const int* hrowE  = hrow ? hrow + (size_t)e * NTOK : nullptr;
  const float* wgtE = wgt  ? wgt  + (size_t)e * NTOK : nullptr;

  const int tid  = threadIdx.x;
  const int wave = tid >> 5;
  const int lane = tid & 31;
  const int wm = wave & 1;
  const int wn = wave >> 1;
  const int l15 = lane & 15;
  const int koff = (lane >> 4) << 3;

  // staging: 8 chunks per thread (tile 0 = A, tile 1 = W2)
  const bf16_t* gsrc[8];
  int loff[8];
#pragma unroll
  for (int i = 0; i < 8; ++i) {
    int c = i * 256 + tid;
    int tile = c >> 10;
    int r = (c >> 3) & 127;
    int ck = c & 7;
    const bf16_t* base;
    if (tile == 0) {
      int lr = rowBase + r;
      int ar;
      if (hrowE) ar = (lr < rows) ? hrowE[lr] : 0;
      else       ar = (lr < rows) ? lr : 0;
      base = Hsrc + (size_t)ar * K;
    } else {
      int col = colBlock + r;
      base = w2 + (size_t)col * K;
    }
    gsrc[i] = base + ck * 8;
    loff[i] = tile * TILE_E + r * PITCH + ck * 8;
  }

  f32x8 acc[4][2] = {};

#pragma unroll
  for (int i = 0; i < 8; ++i) stage_chunk(gsrc[i], &sm[loff[i]]);

  const int nst = K / KS;
#pragma unroll 1
  for (int s = 0; s < nst; ++s) {
    wait_async();
    __syncthreads();
    if (s + 1 < nst) {
      bf16_t* dst = &sm[((s + 1) & 1) * 2 * TILE_E];
      int k1 = (s + 1) * KS;
#pragma unroll
      for (int i = 0; i < 8; ++i) stage_chunk(gsrc[i] + k1, dst + loff[i]);
    }
    const bf16_t* cur = &sm[(s & 1) * 2 * TILE_E];

#pragma unroll
    for (int kk = 0; kk < KS; kk += 32) {
      bf16x16 af[4];
#pragma unroll
      for (int mi = 0; mi < 4; ++mi) {
        int row = wm * 64 + mi * 16 + l15;
        af[mi] = load_frag(&cur[row * PITCH + kk + koff]);
      }
#pragma unroll
      for (int ni = 0; ni < 2; ++ni) {
        int cl = wn * 32 + ni * 16 + l15;
        bf16x16 bw = load_frag(&cur[TILE_E + cl * PITCH + kk + koff]);
#pragma unroll
        for (int mi = 0; mi < 4; ++mi)
          acc[mi][ni] = wmma_bf16(af[mi], bw, acc[mi][ni]);
      }
    }
  }

  const int rsel = (lane >= 16) ? 8 : 0;
#pragma unroll
  for (int mi = 0; mi < 4; ++mi) {
    int lrbase = rowBase + wm * 64 + mi * 16;
#pragma unroll
    for (int ni = 0; ni < 2; ++ni) {
      int col = colBlock + wn * 32 + ni * 16 + l15;
      float bb = b2[col];
#pragma unroll
      for (int v = 0; v < 8; ++v) {
        int lr = lrbase + v + rsel;
        if (lr < rows) {
          float val = acc[mi][ni][v] + bb;
          int orow;
          if (hrowE) { orow = hrowE[lr]; val *= wgtE[lr]; }
          else       { orow = lr; }
          Out[(size_t)orow * N + col] = val;
        }
      }
    }
  }
}

// ---------------------------------------------------------------------------
// Combine: out[t,:] (holds shared z) += sum_k yr[t*4+k, :]
// ---------------------------------------------------------------------------
__global__ void combine_kernel(const float* __restrict__ yr,
                               float* __restrict__ out) {
  int i = blockIdx.x * blockDim.x + threadIdx.x;
  int t = i / DIM;
  int d = i - t * DIM;
  float s = out[i];
#pragma unroll
  for (int k = 0; k < TOPK; ++k)
    s += yr[((size_t)t * TOPK + k) * DIM + d];
  out[i] = s;
}

// ---------------------------------------------------------------------------
extern "C" void kernel_launch(void* const* d_in, const int* in_sizes, int n_in,
                              void* d_out, int out_size, void* d_ws, size_t ws_size,
                              hipStream_t stream) {
  const float* x   = (const float*)d_in[0];
  const float* gw  = (const float*)d_in[1];
  const float* ew1 = (const float*)d_in[2];
  const float* eb1 = (const float*)d_in[3];
  const float* ew2 = (const float*)d_in[4];
  const float* eb2 = (const float*)d_in[5];
  const float* ew3 = (const float*)d_in[6];
  const float* eb3 = (const float*)d_in[7];
  const float* sw1 = (const float*)d_in[8];
  const float* sb1 = (const float*)d_in[9];
  const float* sw2 = (const float*)d_in[10];
  const float* sb2 = (const float*)d_in[11];
  const float* sw3 = (const float*)d_in[12];
  const float* sb3 = (const float*)d_in[13];
  float* out = (float*)d_out;

  size_t off = 0;
  char* base = (char*)d_ws;
  auto carve = [&](size_t bytes) -> void* {
    void* p = base + off;
    off = (off + bytes + 255) & ~(size_t)255;
    return p;
  };
  bf16_t* xb   = (bf16_t*)carve((size_t)NTOK * DIM * 2);
  bf16_t* ew1b = (bf16_t*)carve((size_t)NEXP * INTER * DIM * 2);
  bf16_t* ew3b = (bf16_t*)carve((size_t)NEXP * INTER * DIM * 2);
  bf16_t* ew2b = (bf16_t*)carve((size_t)NEXP * DIM * INTER * 2);
  bf16_t* sw1b = (bf16_t*)carve((size_t)SH_INTER * DIM * 2);
  bf16_t* sw3b = (bf16_t*)carve((size_t)SH_INTER * DIM * 2);
  bf16_t* sw2b = (bf16_t*)carve((size_t)DIM * SH_INTER * 2);
  bf16_t* h_r  = (bf16_t*)carve((size_t)NTOK * TOPK * INTER * 2);
  bf16_t* hs   = (bf16_t*)carve((size_t)NTOK * SH_INTER * 2);
  float*  yr   = (float*) carve((size_t)NTOK * TOPK * DIM * 4);
  int*    cnt  = (int*)   carve(NEXP * 4);
  int*    tok  = (int*)   carve((size_t)NEXP * NTOK * 4);
  int*    hrow = (int*)   carve((size_t)NEXP * NTOK * 4);
  float*  wgt  = (float*) carve((size_t)NEXP * NTOK * 4);
  (void)ws_size; (void)in_sizes; (void)n_in; (void)out_size;

  (void)hipMemsetAsync(cnt, 0, NEXP * sizeof(int), stream);

  auto cvt = [&](const float* s, bf16_t* d, size_t n) {
    int blocks = (int)((n / 4 + 255) / 256);
    cvt_bf16_kernel<<<blocks, 256, 0, stream>>>(s, d, (int)n);
  };
  cvt(x,   xb,   (size_t)NTOK * DIM);
  cvt(ew1, ew1b, (size_t)NEXP * INTER * DIM);
  cvt(ew3, ew3b, (size_t)NEXP * INTER * DIM);
  cvt(ew2, ew2b, (size_t)NEXP * DIM * INTER);
  cvt(sw1, sw1b, (size_t)SH_INTER * DIM);
  cvt(sw3, sw3b, (size_t)SH_INTER * DIM);
  cvt(sw2, sw2b, (size_t)DIM * SH_INTER);

  gate_topk_kernel<<<NTOK / 4, 128, 0, stream>>>(x, gw, cnt, tok, hrow, wgt);

  {  // shared expert gate/up
    dim3 g(SH_INTER / 128, NTOK / 128, 1);
    gemm_gate_up_kernel<<<g, 256, 0, stream>>>(
        xb, sw1b, sw3b, sb1, sb3, hs, nullptr, nullptr, nullptr,
        SH_INTER, DIM, SH_INTER);
  }
  {  // routed gate/up
    dim3 g(INTER / 128, NTOK / 128, NEXP);
    gemm_gate_up_kernel<<<g, 256, 0, stream>>>(
        xb, ew1b, ew3b, eb1, eb3, h_r, tok, hrow, cnt,
        INTER, DIM, INTER);
  }
  {  // shared down -> d_out holds z
    dim3 g(DIM / 128, NTOK / 128, 1);
    gemm_down_kernel<<<g, 256, 0, stream>>>(
        hs, sw2b, sb2, out, nullptr, nullptr, nullptr, DIM, SH_INTER);
  }
  {  // routed down -> yr
    dim3 g(DIM / 128, NTOK / 128, NEXP);
    gemm_down_kernel<<<g, 256, 0, stream>>>(
        h_r, ew2b, eb2, yr, hrow, wgt, cnt, DIM, INTER);
  }
  combine_kernel<<<(NTOK * DIM) / 256, 256, 0, stream>>>(yr, out);
}
